// LSM_partition_cross_partition_inh_12627203850876
// MI455X (gfx1250) — matrix operand order repaired and checked by reading, not compiled, not run on hardware
//
#include <hip/hip_runtime.h>

// ---------------- problem constants (match reference) ----------------
#define T_    512
#define B_    64
#define IN_   512
#define N_    2048
#define P_    4
#define S_    128          // partition steps (long-range delay)
#define R_    129          // spike ring slots (S+1 so read/write slots never collide)
#define ALPHA_ 0.9f
#define BETA_  0.9f
#define TH_    20.0f

typedef __attribute__((ext_vector_type(16))) _Float16 v16h;
typedef __attribute__((ext_vector_type(8)))  _Float16 v8h;
typedef __attribute__((ext_vector_type(8)))  float    v8f;

// ---------------- fragment loaders (CDNA5 wave32 WMMA layouts) ----------------
// A-matrix 16x32 f16: lane = (M = lane&15, half = lane>>4)
//   elements 0..15 -> K = 16*half + e (contiguous, matches B)
static __device__ __forceinline__ v16h load_a_frag(const _Float16* __restrict__ prow, int half) {
  return *(const v16h*)(prow + 16 * half);
}

// B-matrix 32x16 f16: lane = (N = lane&15, half = lane>>4), K = 16*half + e (contiguous).
// We read B[k][n] = W[n][k] straight out of the row-major weight matrix -> no transpose.
static __device__ __forceinline__ v16h load_b_frag(const _Float16* __restrict__ p) {
  return *(const v16h*)p;
}

static __device__ __forceinline__ v8f wmma16(v16h a, v16h b, v8f c) {
  // (neg_a, A, neg_b, B, c_mod, C, reuse_a, reuse_b)
  return __builtin_amdgcn_wmma_f32_16x16x32_f16(false, a, false, b, (short)0, c, false, false);
}

// ---------------- f32 -> f16 bulk convert ----------------
__global__ void __launch_bounds__(256)
f32_to_f16_kernel(const float* __restrict__ src, _Float16* __restrict__ dst, int n) {
  int i = blockIdx.x * blockDim.x + threadIdx.x;
  int stride = gridDim.x * blockDim.x;
  for (; i < n; i += stride) dst[i] = (_Float16)src[i];
}

// ---------------- one recurrence step ----------------
// grid: 32 blocks x 128 threads (4 waves). wave -> 16 output columns, 64 rows (4 accs).
__global__ void __launch_bounds__(128)
lsm_step_kernel(const _Float16* __restrict__ x16,     // [T,B,IN]  f16
                const _Float16* __restrict__ Wins16,  // [P,N,IN]  f16
                const _Float16* __restrict__ Wrec16,  // [N,N]     f16
                const _Float16* __restrict__ Wlong16, // [N,N]     f16
                const float* __restrict__ b1,
                const float* __restrict__ b_long,
                const float* __restrict__ b_rec,
                _Float16* __restrict__ ring,          // [R,B,N]   f16 spike history
                float* __restrict__ syn,              // [B,N]
                float* __restrict__ mem,              // [B,N]
                float* __restrict__ out,              // [T,B,N]
                int t) {
  const int tid  = threadIdx.x;
  const int wave = tid >> 5;
  const int lane = tid & 31;
  const int ln   = lane & 15;
  const int half = lane >> 4;
  const int ncol = blockIdx.x * 64 + wave * 16 + ln;  // this lane's output column

  v8f acc[4] = {};

  const _Float16* __restrict__ ring_prev =
      ring + (size_t)((t + R_ - 1) % R_) * (B_ * N_);

  // ---- spk[t-1] @ Wrec^T ----
  if (t > 0) {
    const _Float16* __restrict__ Br = Wrec16 + (size_t)ncol * N_ + 16 * half;
#pragma unroll 2
    for (int k0 = 0; k0 < N_; k0 += 32) {
      v16h bf = load_b_frag(Br + k0);
#pragma unroll
      for (int m = 0; m < 4; ++m) {
        const _Float16* pa = ring_prev + (size_t)(16 * m + ln) * N_ + k0;
        acc[m] = wmma16(load_a_frag(pa, half), bf, acc[m]);
      }
    }
  }

  // ---- spk[t-S] @ Wlong^T  (inh_mask: strictly t > S) ----
  if (t > S_) {
    const _Float16* __restrict__ Adel =
        ring + (size_t)((t - S_) % R_) * (B_ * N_);
    const _Float16* __restrict__ Br = Wlong16 + (size_t)ncol * N_ + 16 * half;
#pragma unroll 2
    for (int k0 = 0; k0 < N_; k0 += 32) {
      v16h bf = load_b_frag(Br + k0);
#pragma unroll
      for (int m = 0; m < 4; ++m) {
        const _Float16* pa = Adel + (size_t)(16 * m + ln) * N_ + k0;
        acc[m] = wmma16(load_a_frag(pa, half), bf, acc[m]);
      }
    }
  }

  // ---- x[t] @ Wins[p]^T ----
  {
    const int p = t / S_;
    const _Float16* __restrict__ Ax = x16 + (size_t)t * (B_ * IN_);
    const _Float16* __restrict__ Br =
        Wins16 + ((size_t)p * N_ + ncol) * IN_ + 16 * half;
#pragma unroll 2
    for (int k0 = 0; k0 < IN_; k0 += 32) {
      v16h bf = load_b_frag(Br + k0);
#pragma unroll
      for (int m = 0; m < 4; ++m) {
        const _Float16* pa = Ax + (size_t)(16 * m + ln) * IN_ + k0;
        acc[m] = wmma16(load_a_frag(pa, half), bf, acc[m]);
      }
    }
  }

  // ---- fused RSynaptic epilogue ----
  // C/D layout: lanes 0-15 VGPR r -> M=r, lanes 16-31 VGPR r -> M=r+8; N = lane&15.
  const float biasn = b1[ncol] + b_rec[ncol] + ((t > S_) ? b_long[ncol] : 0.0f);
  _Float16* __restrict__ ring_cur = ring + (size_t)(t % R_) * (B_ * N_);
  float* __restrict__ outt = out + (size_t)t * (B_ * N_);

#pragma unroll
  for (int m = 0; m < 4; ++m) {
#pragma unroll
    for (int r = 0; r < 8; ++r) {
      const int brow = 16 * m + r + 8 * half;
      const size_t idx = (size_t)brow * N_ + ncol;
      const float sp = (t > 0) ? (float)ring_prev[idx] : 0.0f;  // reset source
      const float s  = ALPHA_ * syn[idx] + acc[m][r] + biasn;
      const float mv = BETA_ * mem[idx] + s - sp * TH_;
      const float spk = (mv > TH_) ? 1.0f : 0.0f;
      syn[idx] = s;
      mem[idx] = mv;
      outt[idx] = spk;
      ring_cur[idx] = (_Float16)spk;
    }
  }
}

// ---------------- host-side launch ----------------
extern "C" void kernel_launch(void* const* d_in, const int* in_sizes, int n_in,
                              void* d_out, int out_size, void* d_ws, size_t ws_size,
                              hipStream_t stream) {
  const float* x      = (const float*)d_in[0];  // [T,B,IN]
  const float* Wins   = (const float*)d_in[1];  // [P,N,IN]
  const float* b1     = (const float*)d_in[2];  // [N]
  const float* Wlong  = (const float*)d_in[3];  // [N,N]
  const float* b_long = (const float*)d_in[4];  // [N]
  const float* Wrec   = (const float*)d_in[5];  // [N,N]
  const float* b_rec  = (const float*)d_in[6];  // [N]
  float* out = (float*)d_out;                   // [T,B,N]

  // workspace carve-up (256B aligned)
  char* ws = (char*)d_ws;
  size_t off = 0;
  auto carve = [&](size_t bytes) -> void* {
    void* p = ws + off;
    off = (off + bytes + 255) & ~(size_t)255;
    return p;
  };
  const size_t n_x16    = (size_t)T_ * B_ * IN_;   // 16.8M
  const size_t n_wins   = (size_t)P_ * N_ * IN_;   // 4.2M
  const size_t n_wsq    = (size_t)N_ * N_;         // 4.2M
  const size_t n_ring   = (size_t)R_ * B_ * N_;    // 16.9M
  const size_t n_state  = (size_t)B_ * N_;         // 131K

  _Float16* x16     = (_Float16*)carve(n_x16 * sizeof(_Float16));
  _Float16* Wins16  = (_Float16*)carve(n_wins * sizeof(_Float16));
  _Float16* Wrec16  = (_Float16*)carve(n_wsq * sizeof(_Float16));
  _Float16* Wlong16 = (_Float16*)carve(n_wsq * sizeof(_Float16));
  _Float16* ring    = (_Float16*)carve(n_ring * sizeof(_Float16));
  float*    syn     = (float*)carve(n_state * sizeof(float));
  float*    memb    = (float*)carve(n_state * sizeof(float));

  // zero the recurrent state (graph-capture safe)
  hipMemsetAsync(syn,  0, n_state * sizeof(float), stream);
  hipMemsetAsync(memb, 0, n_state * sizeof(float), stream);

  // one-time (per call) f16 conversions; weights then live in L2 for all 512 steps
  f32_to_f16_kernel<<<2048, 256, 0, stream>>>(x,     x16,     (int)n_x16);
  f32_to_f16_kernel<<<2048, 256, 0, stream>>>(Wins,  Wins16,  (int)n_wins);
  f32_to_f16_kernel<<<2048, 256, 0, stream>>>(Wrec,  Wrec16,  (int)n_wsq);
  f32_to_f16_kernel<<<2048, 256, 0, stream>>>(Wlong, Wlong16, (int)n_wsq);

  // strictly sequential recurrence: one launch per timestep, serialized on `stream`
  for (int t = 0; t < T_; ++t) {
    lsm_step_kernel<<<32, 128, 0, stream>>>(x16, Wins16, Wrec16, Wlong16,
                                            b1, b_long, b_rec,
                                            ring, syn, memb, out, t);
  }
}
